// MultiQuerySelfAttention_46170898432419
// MI455X (gfx1250) — matrix operand order repaired and checked by reading, not compiled, not run on hardware
//
#include <hip/hip_runtime.h>
#include <hip/hip_bf16.h>

typedef __bf16 bf16;
typedef __attribute__((ext_vector_type(16))) __bf16 v16bf;
typedef __attribute__((ext_vector_type(8)))  float  v8f;

// ---------- helpers ----------

__device__ inline bf16 f2bf(float f) {
  unsigned u; __builtin_memcpy(&u, &f, 4);
  u += 0x7FFFu + ((u >> 16) & 1u);            // round-to-nearest-even
  unsigned short hs = (unsigned short)(u >> 16);
  bf16 r; __builtin_memcpy(&r, &hs, 2);
  return r;
}

// A-fragment, 16x32 bf16 (ISA 7.12.2): lane row = lane%16, half h = lane/16.
// elements 0..7  -> K = k0 + 8h + e ; elements 8..15 -> K = k0 + 16 + 8h + (e-8)
__device__ inline v16bf afrag_bf16(const bf16* tile, int ld, int r, int h, int k0) {
  const bf16* p = tile + r * ld + k0 + 8 * h;
  uint4 lo = *reinterpret_cast<const uint4*>(p);
  uint4 hi = *reinterpret_cast<const uint4*>(p + 16);
  v16bf out;
  __builtin_memcpy(&out, &lo, 16);
  __builtin_memcpy(reinterpret_cast<char*>(&out) + 16, &hi, 16);
  return out;
}
// B-fragment, 32x16 bf16: lane col n = lane%16; elements 0..7 -> K = k0 + 8h + e ; 8..15 -> K = k0 + 16 + 8h + (e-8)
__device__ inline v16bf bfrag_bf16(const bf16* tile, int ld, int n, int h, int k0) {
  const bf16* p = tile + n * ld + k0 + 8 * h;
  uint4 lo = *reinterpret_cast<const uint4*>(p);
  uint4 hi = *reinterpret_cast<const uint4*>(p + 16);
  v16bf out;
  __builtin_memcpy(&out, &lo, 16);
  __builtin_memcpy(reinterpret_cast<char*>(&out) + 16, &hi, 16);
  return out;
}

__device__ inline v8f wmma_bf16(v16bf a, v16bf b, v8f c) {
  return __builtin_amdgcn_wmma_f32_16x16x32_bf16(false, a, false, b, (short)0, c, false, false);
}

// async global -> LDS stage, 16 bytes per lane (ASYNCcnt-tracked on gfx1250)
__device__ inline void stage16(const bf16* g, bf16* l) {
  unsigned lds_off =
      (unsigned)(unsigned long long)(__attribute__((address_space(3))) bf16*)l;
  asm volatile("global_load_async_to_lds_b128 %0, %1, off"
               :
               : "v"(lds_off), "v"(g)
               : "memory");
}
__device__ inline void wait_async() {
  asm volatile("s_wait_asynccnt 0" ::: "memory");
}

// reductions across the 16 lanes of a half-wave (halves stay independent)
__device__ inline float rowmax16(float v) {
#pragma unroll
  for (int m = 8; m >= 1; m >>= 1) v = fmaxf(v, __shfl_xor(v, m, 32));
  return v;
}
__device__ inline float rowsum16(float v) {
#pragma unroll
  for (int m = 8; m >= 1; m >>= 1) v += __shfl_xor(v, m, 32);
  return v;
}

// ---------- kernel 0: one-time fp32 -> bf16 packing ----------

__global__ __launch_bounds__(256) void cvt_bf16(const float* __restrict__ in,
                                                bf16* __restrict__ out, int n4) {
  const int i = blockIdx.x * 256 + threadIdx.x;
  if (i < n4) {
    float4 v = reinterpret_cast<const float4*>(in)[i];
    bf16 o[4] = {f2bf(v.x), f2bf(v.y), f2bf(v.z), f2bf(v.w)};
    uint2 pk; __builtin_memcpy(&pk, o, 8);
    reinterpret_cast<uint2*>(out)[i] = pk;
  }
}

// ---------- kernel 1: C = A(bf16, MxK) . B(bf16, NxK)^T, f32 out ----------
// Block: 8 waves x (32x64) tiles sharing one B slab; B staged to LDS with
// async copies, double-buffered; K = 1024 in 32 slabs of 32.

__global__ __launch_bounds__(256) void gemm_bf16(const bf16* __restrict__ A,
                                                 const bf16* __restrict__ B,
                                                 float* __restrict__ C, int ldc) {
  __shared__ bf16 Bs[2][64][32];                     // 2 x 4KB: [buf][n][k]
  const int tid = threadIdx.x;
  const int lane = tid & 31, wave = tid >> 5;
  const int rl = lane & 15, h = lane >> 4;
  const int row0 = (blockIdx.x * 8 + wave) * 32;
  const int n0 = blockIdx.y * 64;

  // cooperative B staging: thread -> (row nr, 16B segment seg) of the 64x32 slab
  const int nr = tid >> 2, seg = tid & 3;
  const bf16* gB = B + (n0 + nr) * 1024 + seg * 8;

  v8f acc[2][4];
#pragma unroll
  for (int i = 0; i < 2; ++i)
#pragma unroll
    for (int j = 0; j < 4; ++j)
#pragma unroll
      for (int r = 0; r < 8; ++r) acc[i][j][r] = 0.f;

  const bf16* A0 = A + row0 * 1024;
  const bf16* A1 = A + (row0 + 16) * 1024;

  stage16(gB, &Bs[0][nr][seg * 8]);                  // prologue: slab 0 -> buf 0
  for (int s = 0; s < 32; ++s) {
    const int k0 = s * 32;
    const int buf = s & 1;
    wait_async();                                    // my slab-s writes landed
    __syncthreads();                                 // everyone's landed; buf^1 readers done
    if (s + 1 < 32) stage16(gB + k0 + 32, &Bs[buf ^ 1][nr][seg * 8]);

    v16bf a0 = afrag_bf16(A0, 1024, rl, h, k0);
    v16bf a1 = afrag_bf16(A1, 1024, rl, h, k0);
#pragma unroll
    for (int j = 0; j < 4; ++j) {
      v16bf b = bfrag_bf16(&Bs[buf][16 * j][0], 32, rl, h, 0);
      acc[0][j] = wmma_bf16(a0, b, acc[0][j]);
      acc[1][j] = wmma_bf16(a1, b, acc[1][j]);
    }
  }
#pragma unroll
  for (int i = 0; i < 2; ++i)
#pragma unroll
    for (int j = 0; j < 4; ++j)
#pragma unroll
      for (int r = 0; r < 8; ++r)
        C[(row0 + 16 * i + r + 8 * h) * ldc + n0 + 16 * j + rl] = acc[i][j][r];
}

// ---------- kernel 2: LayerNorm + split into K (row-major), V^T, Q ----------

__global__ __launch_bounds__(128) void ln_split(const float* __restrict__ qkv,
                                                const float* __restrict__ gamma,
                                                const float* __restrict__ beta,
                                                bf16* __restrict__ Kmat,   // [4096][64]
                                                bf16* __restrict__ Vt,     // [2][64][2048]
                                                bf16* __restrict__ Qmat) { // [4096][1024]
  __shared__ float s1[128], s2[128];
  const int row = blockIdx.x;
  const int tid = threadIdx.x;
  const float* p = qkv + row * 1152;
  float sum = 0.f, sq = 0.f;
  for (int j = tid; j < 1152; j += 128) { float v = p[j]; sum += v; sq += v * v; }
  s1[tid] = sum; s2[tid] = sq; __syncthreads();
  for (int s = 64; s > 0; s >>= 1) {
    if (tid < s) { s1[tid] += s1[tid + s]; s2[tid] += s2[tid + s]; }
    __syncthreads();
  }
  const float mu = s1[0] * (1.f / 1152.f);
  const float var = s2[0] * (1.f / 1152.f) - mu * mu;
  const float rstd = rsqrtf(var + 1e-5f);
  const int b = row >> 11, l = row & 2047;
  for (int j = tid; j < 1152; j += 128) {
    bf16 yb = f2bf((p[j] - mu) * rstd * gamma[j] + beta[j]);
    if (j < 64)       Kmat[row * 64 + j] = yb;
    else if (j < 128) Vt[(b * 64 + (j - 64)) * 2048 + l] = yb;
    else              Qmat[row * 1024 + (j - 128)] = yb;
  }
}

// ---------- kernel 3: flash attention, one wave per (batch, head, 16-query tile) ----------

__global__ __launch_bounds__(256) void flash_attn(const bf16* __restrict__ Qmat,
                                                  const bf16* __restrict__ Kmat,
                                                  const bf16* __restrict__ Vt,
                                                  bf16* __restrict__ Omat) {
  __shared__ bf16 pshare[8][16][32];               // per-wave P tile (C-layout -> A-layout)
  const int lane = threadIdx.x & 31;
  const int wave = threadIdx.x >> 5;
  const int rl = lane & 15, h = lane >> 4;
  const int pair = blockIdx.x * 8 + wave;          // 32 (batch,head) pairs
  const int b = pair >> 4, head = pair & 15;
  const int qbase = blockIdx.y * 16;               // all waves in block share qbase
  const float NEG_INF = -__builtin_huge_valf();
  const float scale = 0.125f;                      // 1/sqrt(64)

  const bf16* Qtile = Qmat + (b * 2048 + qbase) * 1024 + head * 64;
  v16bf aq0 = afrag_bf16(Qtile, 1024, rl, h, 0);
  v16bf aq1 = afrag_bf16(Qtile, 1024, rl, h, 32);

  float mi[8], li[8];
  v8f oacc[4];
#pragma unroll
  for (int r = 0; r < 8; ++r) {
    mi[r] = NEG_INF; li[r] = 0.f;
#pragma unroll
    for (int t = 0; t < 4; ++t) oacc[t][r] = 0.f;
  }

  const int kend = qbase + 16;                     // exclusive causal key bound
  for (int kc = 0; kc < kend; kc += 32) {
    // --- scores: Q (16x64) . K^T -> two 16x16 tiles covering keys [kc, kc+32)
    v8f s0, s1;
#pragma unroll
    for (int r = 0; r < 8; ++r) { s0[r] = 0.f; s1[r] = 0.f; }
    {
      const bf16* kt = Kmat + (b * 2048 + kc) * 64;
      s0 = wmma_bf16(aq0, bfrag_bf16(kt, 64, rl, h, 0),  s0);
      s0 = wmma_bf16(aq1, bfrag_bf16(kt, 64, rl, h, 32), s0);
    }
    {
      const bf16* kt = Kmat + (b * 2048 + kc + 16) * 64;
      s1 = wmma_bf16(aq0, bfrag_bf16(kt, 64, rl, h, 0),  s1);
      s1 = wmma_bf16(aq1, bfrag_bf16(kt, 64, rl, h, 32), s1);
    }
    // --- online softmax in C-layout: row m = r + 8h, col n = rl
#pragma unroll
    for (int r = 0; r < 8; ++r) {
      const int q = qbase + r + 8 * h;
      float v0 = s0[r] * scale; if (kc + rl      > q) v0 = NEG_INF;
      float v1 = s1[r] * scale; if (kc + 16 + rl > q) v1 = NEG_INF;
      const float rm = fmaxf(rowmax16(fmaxf(v0, v1)), mi[r]);
      const float c = __expf(mi[r] - rm);
      const float p0 = __expf(v0 - rm);
      const float p1 = __expf(v1 - rm);
      li[r] = li[r] * c + rowsum16(p0 + p1);
      mi[r] = rm;
#pragma unroll
      for (int t = 0; t < 4; ++t) oacc[t][r] *= c;
      pshare[wave][r + 8 * h][rl]      = f2bf(p0);
      pshare[wave][r + 8 * h][16 + rl] = f2bf(p1);
    }
    __syncthreads();
    // --- P (16x32) . V (32x64): V^T rows are key-contiguous
    v16bf ap = afrag_bf16(&pshare[wave][0][0], 32, rl, h, 0);
#pragma unroll
    for (int t = 0; t < 4; ++t) {
      v16bf bv = bfrag_bf16(Vt + (b * 64 + 16 * t) * 2048, 2048, rl, h, kc);
      oacc[t] = wmma_bf16(ap, bv, oacc[t]);
    }
    __syncthreads();
  }

  bf16* O = Omat + (b * 2048 + qbase) * 1024 + head * 64;
#pragma unroll
  for (int r = 0; r < 8; ++r) {
    const float inv = 1.f / li[r];
#pragma unroll
    for (int t = 0; t < 4; ++t)
      O[(r + 8 * h) * 1024 + 16 * t + rl] = f2bf(oacc[t][r] * inv);
  }
}

// ---------- launch ----------

extern "C" void kernel_launch(void* const* d_in, const int* in_sizes, int n_in,
                              void* d_out, int out_size, void* d_ws, size_t ws_size,
                              hipStream_t stream) {
  const float* x     = (const float*)d_in[0];
  const float* Wqkv  = (const float*)d_in[1];
  const float* gamma = (const float*)d_in[2];
  const float* beta  = (const float*)d_in[3];
  const float* Wfc   = (const float*)d_in[4];
  float* out = (float*)d_out;

  char* ws = (char*)d_ws;
  float* qkv  = (float*)(ws);                          // 4096*1152*4 = 18,874,368
  bf16* Kmat  = (bf16*)(ws + 18874368);                // 4096*64*2   =    524,288
  bf16* Vt    = (bf16*)(ws + 18874368 + 524288);       // 2*64*2048*2 =    524,288
  bf16* Qmat  = (bf16*)(ws + 19922944);                // 4096*1024*2 =  8,388,608
  bf16* Omat  = (bf16*)(ws + 28311552);                // 4096*1024*2 =  8,388,608
  bf16* xb    = (bf16*)(ws + 36700160);                // 4096*1024*2 =  8,388,608
  bf16* Wqkvb = (bf16*)(ws + 45088768);                // 1152*1024*2 =  2,359,296
  bf16* Wfcb  = (bf16*)(ws + 47448064);                // 1024*1024*2 =  2,097,152

  // one-time bf16 packing (element counts / 4)
  cvt_bf16<<<dim3(1048576 / 256), 256, 0, stream>>>(x,    xb,    1048576);
  cvt_bf16<<<dim3(294912  / 256), 256, 0, stream>>>(Wqkv, Wqkvb, 294912);
  cvt_bf16<<<dim3(262144  / 256), 256, 0, stream>>>(Wfc,  Wfcb,  262144);

  gemm_bf16 <<<dim3(16, 18), 256, 0, stream>>>(xb, Wqkvb, qkv, 1152);
  ln_split  <<<dim3(4096),   128, 0, stream>>>(qkv, gamma, beta, Kmat, Vt, Qmat);
  flash_attn<<<dim3(4, 128), 256, 0, stream>>>(Qmat, Kmat, Vt, Omat);
  gemm_bf16 <<<dim3(16, 16), 256, 0, stream>>>(Omat, Wfcb, out, 1024);
}
